// LogLSTM_71957882077828
// MI455X (gfx1250) — matrix operand order repaired and checked by reading, not compiled, not run on hardware
//
#include <hip/hip_runtime.h>
#include <hip/hip_bf16.h>
#include <math.h>

#define VOCAB  50000
#define E_DIM  128
#define H_DIM  128
#define B_SZ   256
#define T_LEN  512
#define G4H    512            // 4*H

typedef __attribute__((ext_vector_type(16))) __bf16 v16bf;
typedef __attribute__((ext_vector_type(8)))  __bf16 v8bf;
typedef __attribute__((ext_vector_type(4)))  __bf16 v4bf;
typedef __attribute__((ext_vector_type(8)))  float  v8f;

// ---- WMMA fragment helpers (wave32; contiguous K order k = 16*half + s, used alike for A and B) ----
// A 16x32 bf16: lane<16 holds row M=lane, K = kb+0..15;
//               lane>=16 holds same row, K = kb+16..31.
static __device__ inline v16bf a_frag_bf16(const __bf16* row, int kb, int half) {
  const __bf16* p = row + kb + half * 16;
  v8bf lo = *(const v8bf*)p;          // K = k0 .. k0+7
  v8bf hi = *(const v8bf*)(p + 8);    // K = k0+8 .. k0+15
  return __builtin_shufflevector(lo, hi, 0,1,2,3,4,5,6,7,8,9,10,11,12,13,14,15);
}
// B 32x16 bf16: lane<16 holds column N=lane, K = kb+0..15; lane>=16 holds N=lane-16,
// K = kb+16..31. W is row-major [N][128] so each lane reads 16 contiguous bf16 (32 B).
static __device__ inline v16bf b_frag_bf16(const __bf16* W, int n, int kb, int half) {
  return *(const v16bf*)(W + (size_t)n * 128 + kb + half * 16);
}
static __device__ inline v8f wmma_bf16(v16bf a, v16bf b, v8f c) {
  return __builtin_amdgcn_wmma_f32_16x16x32_bf16(false, a, false, b, (short)0, c, false, false);
}
// Transcendental-unit activations (v_exp_f32 + v_rcp_f32; co-execute with XDL WMMA)
static __device__ inline float fast_sig(float x)  {
  return __builtin_amdgcn_rcpf(1.0f + __expf(-x));
}
static __device__ inline float fast_tanh(float x) {
  return 2.0f * __builtin_amdgcn_rcpf(1.0f + __expf(-2.0f * x)) - 1.0f;
}

// ---------------------------------------------------------------------------
// Kernel 0: fp32 -> bf16 bulk conversion (4 elements/thread, v_cvt_pk_bf16_f32)
// Used once for emb_table (25.6MB->12.8MB), W_ih, W_hh. Removes all per-use
// conversion VALU from the GEMM inner loops and halves gather traffic.
// ---------------------------------------------------------------------------
__global__ void cvt_bf16x4_kernel(const float* __restrict__ in, __bf16* __restrict__ out, int n4) {
  int i = blockIdx.x * blockDim.x + threadIdx.x;
  if (i < n4) {
    float4 v = ((const float4*)in)[i];
    v4bf o = { (__bf16)v.x, (__bf16)v.y, (__bf16)v.z, (__bf16)v.w };
    ((v4bf*)out)[i] = o;
  }
}

// ---------------------------------------------------------------------------
// Kernel 1: fused embedding gather + input projection GEMM (bf16 WMMA, f32 acc)
//   xp[t][n][b] = sum_k emb[x[b,t]][k] * W_ih[n][k] + b_ih[n] + b_hh[n]   (bf16)
// Row m = b*T + t. Grid: 8192 blocks x 256 threads (8 waves); block = 128 rows
// x 64 cols, wave = 16 rows x 64 cols (4 WMMA tiles, K-loop of 4). Inner loop
// is pure bf16 global_load_b128 -> v_wmma (no conversions).
// ---------------------------------------------------------------------------
__global__ __launch_bounds__(256)
void embed_proj_kernel(const int* __restrict__ x, const __bf16* __restrict__ emb_bf,
                       const __bf16* __restrict__ wih_bf, const float* __restrict__ b_ih,
                       const float* __restrict__ b_hh, __bf16* __restrict__ xp) {
  const int wave = threadIdx.x >> 5, lane = threadIdx.x & 31;
  const int half = lane >> 4, l16 = lane & 15;
  const int bm = blockIdx.x >> 3, bn = blockIdx.x & 7;
  const int m0 = bm * 128 + wave * 16;
  const int n0 = bn * 64;

  const int mrow = m0 + l16;                  // both lane halves share row M = l16
  const int idx  = x[mrow];                   // embedding gather (padding row 0 is zeros)
  const __bf16* arow = emb_bf + (size_t)idx * E_DIM;

  v8f acc[4] = {};
#pragma unroll
  for (int kk = 0; kk < 4; ++kk) {
    const int kb = kk * 32;
    v16bf a = a_frag_bf16(arow, kb, half);
#pragma unroll
    for (int nt = 0; nt < 4; ++nt) {
      v16bf b = b_frag_bf16(wih_bf, n0 + nt * 16 + l16, kb, half);
      acc[nt] = wmma_bf16(a, b, acc[nt]);
    }
  }
  // C layout: VGPR r -> M = m0 + r + 8*half, N = n0 + nt*16 + l16
#pragma unroll
  for (int nt = 0; nt < 4; ++nt) {
    const int n = n0 + nt * 16 + l16;
    const float bias = b_ih[n] + b_hh[n];
#pragma unroll
    for (int r = 0; r < 8; ++r) {
      const int m = m0 + r + 8 * half;
      const int bidx = m >> 9;                // m / T
      const int t    = m & (T_LEN - 1);       // m % T
      xp[(size_t)t * (G4H * B_SZ) + (size_t)n * B_SZ + bidx] = (__bf16)(acc[nt][r] + bias);
    }
  }
}

// ---------------------------------------------------------------------------
// Kernel 2: persistent single-workgroup LSTM recurrence + fused classifier.
// 512 threads = 16 waves on one WGP. h double-buffered in LDS (bf16, 2x64KB),
// c persistent in VGPRs. Wave grid 4x4: wave (mi,ji) owns M-tiles mi*4..mi*4+3
// and H-chunks j = ji*2, ji*2+1 WITH all four gates (i,f,g,o) -> local update.
// Per step/wave: 128 v_wmma (A from LDS, B from L2-resident W_hh bf16 so the
// LDS and L2 pipes split the fragment traffic), TRANS-unit gate math, 1 barrier.
// ---------------------------------------------------------------------------
__global__ __launch_bounds__(512)
void lstm_rec_kernel(const __bf16* __restrict__ xp, const __bf16* __restrict__ whh,
                     const float* __restrict__ W_c, const float* __restrict__ b_c,
                     float* __restrict__ out) {
  extern __shared__ __bf16 smem[];
  __bf16* h0 = smem;                   // [256][128]
  __bf16* h1 = smem + B_SZ * H_DIM;    // [256][128]

  const int tid  = threadIdx.x;
  const int wave = tid >> 5, lane = tid & 31;
  const int half = lane >> 4, l16 = lane & 15;
  const int mi = wave >> 2, ji = wave & 3;

  for (int i = tid; i < B_SZ * H_DIM; i += 512) h0[i] = (__bf16)0.0f;   // h(0) = 0
  __syncthreads();

  float c_reg[8][8];                   // c(0) = 0, persistent across all 512 steps
#pragma unroll
  for (int q = 0; q < 8; ++q)
#pragma unroll
    for (int r = 0; r < 8; ++r) c_reg[q][r] = 0.0f;

  for (int t = 0; t < T_LEN; ++t) {
    const __bf16* cur = (t & 1) ? h1 : h0;
    __bf16*       nxt = (t & 1) ? h0 : h1;
    const __bf16* xpt = xp + (size_t)t * (G4H * B_SZ);
    if (t + 1 < T_LEN)                 // gfx1250 global_prefetch_b8 on next step's slice
      __builtin_prefetch(xp + (size_t)(t + 1) * (G4H * B_SZ) + (size_t)tid * 256, 0, 1);

#pragma unroll
    for (int mm = 0; mm < 4; ++mm) {
      const int mt = mi * 4 + mm;
      const __bf16* hrow = cur + (size_t)(mt * 16 + l16) * H_DIM;
      v16bf afr[4];
#pragma unroll
      for (int kk = 0; kk < 4; ++kk) afr[kk] = a_frag_bf16(hrow, kk * 32, half);

#pragma unroll
      for (int jj = 0; jj < 2; ++jj) {
        const int j = ji * 2 + jj;
        v8f acc[4] = {};
#pragma unroll
        for (int g = 0; g < 4; ++g) {                 // gate order i,f,g,o
          const int n = g * 128 + j * 16 + l16;
#pragma unroll
          for (int kk = 0; kk < 4; ++kk)
            acc[g] = wmma_bf16(afr[kk], b_frag_bf16(whh, n, kk * 32, half), acc[g]);
        }
        // x-projection: layout xp[t][n][b] makes each fragment one contiguous v8bf
        const int bbase = mt * 16 + 8 * half;
        v8bf xv[4];
#pragma unroll
        for (int g = 0; g < 4; ++g) {
          const int n = g * 128 + j * 16 + l16;
          xv[g] = *(const v8bf*)(xpt + (size_t)n * B_SZ + bbase);
        }
        const int q = mm * 2 + jj;
#pragma unroll
        for (int r = 0; r < 8; ++r) {
          float iv = fast_sig (acc[0][r] + (float)xv[0][r]);
          float fv = fast_sig (acc[1][r] + (float)xv[1][r]);
          float gv = fast_tanh(acc[2][r] + (float)xv[2][r]);
          float ov = fast_sig (acc[3][r] + (float)xv[3][r]);
          float cc = fv * c_reg[q][r] + iv * gv;
          c_reg[q][r] = cc;
          nxt[(size_t)(bbase + r) * H_DIM + j * 16 + l16] = (__bf16)(ov * fast_tanh(cc));
        }
      }
    }
    __syncthreads();                   // h(t+1) visible to all waves
  }

  // Final h lives in h0 (written as nxt at t=511). Classifier: 512 threads = 256 b x 2 c.
  {
    const int b = tid >> 1, c = tid & 1;
    const __bf16* hv = h0 + (size_t)b * H_DIM;
    const float*  wv = W_c + (size_t)c * H_DIM;
    float s = 0.0f;
#pragma unroll 8
    for (int k = 0; k < H_DIM; ++k) s += (float)hv[k] * wv[k];
    out[b * 2 + c] = s + b_c[c];
  }
}

// ---------------------------------------------------------------------------
extern "C" void kernel_launch(void* const* d_in, const int* in_sizes, int n_in,
                              void* d_out, int out_size, void* d_ws, size_t ws_size,
                              hipStream_t stream) {
  (void)in_sizes; (void)n_in; (void)out_size; (void)ws_size;
  const int*   x   = (const int*)  d_in[0];
  const float* emb = (const float*)d_in[1];
  const float* Wih = (const float*)d_in[2];
  const float* Whh = (const float*)d_in[3];
  const float* bih = (const float*)d_in[4];
  const float* bhh = (const float*)d_in[5];
  const float* Wc  = (const float*)d_in[6];
  const float* bc  = (const float*)d_in[7];
  float* out = (float*)d_out;

  // Workspace layout (bf16 elements):
  //   xp [T][4H][B] : 67,108,864   (128 MB)
  //   W_hh bf16     :     65,536   (128 KB)
  //   W_ih bf16     :     65,536   (128 KB)
  //   emb bf16      :  6,400,000   (12.8 MB)
  __bf16* xp     = (__bf16*)d_ws;
  __bf16* whh_bf = xp + (size_t)T_LEN * G4H * B_SZ;
  __bf16* wih_bf = whh_bf + (size_t)G4H * H_DIM;
  __bf16* emb_bf = wih_bf + (size_t)G4H * E_DIM;

  const int n4_w   = (G4H * H_DIM) / 4;        // 16384
  const int n4_emb = (VOCAB * E_DIM) / 4;      // 1,600,000
  cvt_bf16x4_kernel<<<(n4_w   + 255) / 256, 256, 0, stream>>>(Whh, whh_bf, n4_w);
  cvt_bf16x4_kernel<<<(n4_w   + 255) / 256, 256, 0, stream>>>(Wih, wih_bf, n4_w);
  cvt_bf16x4_kernel<<<(n4_emb + 255) / 256, 256, 0, stream>>>(emb, emb_bf, n4_emb);

  embed_proj_kernel<<<8192, 256, 0, stream>>>(x, emb_bf, wih_bf, bih, bhh, xp);

  const size_t shmem = (size_t)2 * B_SZ * H_DIM * sizeof(__bf16);   // 128 KB (<320 KB/WGP)
  hipFuncSetAttribute(reinterpret_cast<const void*>(lstm_rec_kernel),
                      hipFuncAttributeMaxDynamicSharedMemorySize, (int)shmem);
  lstm_rec_kernel<<<1, 512, shmem, stream>>>(xp, whh_bf, Wc, bc, out);
}
